// MultiScaleAttention_26645977105168
// MI455X (gfx1250) — matrix operand order repaired and checked by reading, not compiled, not run on hardware
//
#include <hip/hip_runtime.h>
#include <hip/hip_bf16.h>

typedef __attribute__((ext_vector_type(16))) __bf16 bf16x16;
typedef __attribute__((ext_vector_type(8)))  float  f32x8;
typedef __attribute__((ext_vector_type(4)))  unsigned int u32x4;
typedef __attribute__((ext_vector_type(4)))  int i32x4;
typedef __attribute__((ext_vector_type(8)))  int i32x8;
typedef unsigned short u16;
typedef unsigned int   u32;

#define NH   12
#define HD   64
#define SEQ  4096
#define DIM  768
#define QKVN 2304

#if __has_builtin(__builtin_amdgcn_tensor_load_to_lds)
#define ATT_TDM 1
#else
#define ATT_TDM 0
#endif

// ---------------- helpers ----------------

__device__ __forceinline__ u16 f2bf(float f) {
  union { float f; u32 u; } x; x.f = f;
  u32 r = x.u + 0x7FFFu + ((x.u >> 16) & 1u);   // round-to-nearest-even
  return (u16)(r >> 16);
}

// A fragment: 16x32 bf16, contiguous k order (k = 16h + s), same as the B loaders.
// lane L holds row m0+(L&15); VGPR v holds K pair k = kbase + (L>=16)*16 + 2v
__device__ __forceinline__ bf16x16 load_a16(const u16* base, int ld, int m0, int kbase) {
  int lane = threadIdx.x & 31;
  const u16* p = base + (size_t)(m0 + (lane & 15)) * ld + kbase + ((lane >> 4) << 4);
  union { bf16x16 v; u32 u[8]; } r;
#pragma unroll
  for (int v = 0; v < 8; ++v) {
    int k = v << 1;
    r.u[v] = *(const u32*)(p + k);
  }
  return r.v;
}

// Same layout, source in LDS (16 rows x 64 cols bf16 tile)
__device__ __forceinline__ bf16x16 load_a16_lds(const u16* base, int kbase) {
  int lane = threadIdx.x & 31;
  const u16* p = base + (lane & 15) * 64 + kbase + ((lane >> 4) << 4);
  union { bf16x16 v; u32 u[8]; } r;
#pragma unroll
  for (int v = 0; v < 8; ++v) {
    int k = v << 1;
    r.u[v] = *(const u32*)(p + k);
  }
  return r.v;
}

// B fragment 32x16 bf16; memory holds rows indexed by n with K contiguous (N-major, K-contig).
// lane L holds col n0+(L&15); VGPR v holds K pair k = kbase + (L>=16)*16 + 2v
__device__ __forceinline__ bf16x16 load_b16(const u16* base, int ld, int n0, int kbase) {
  int lane = threadIdx.x & 31;
  const u16* p = base + (size_t)(n0 + (lane & 15)) * ld + kbase + ((lane >> 4) << 4);
  union { bf16x16 v; u32 u[8]; } r;
#pragma unroll
  for (int v = 0; v < 8; ++v) r.u[v] = *(const u32*)(p + (v << 1));
  return r.v;
}

// B fragment where memory is K-major: element (k,n) at base[k*ld + n]
__device__ __forceinline__ bf16x16 load_b16_kmajor(const u16* base, int ld, int n0, int kbase) {
  int lane = threadIdx.x & 31;
  int n  = n0 + (lane & 15);
  int k0 = kbase + ((lane >> 4) << 4);
  union { bf16x16 v; u32 u[8]; } r;
#pragma unroll
  for (int v = 0; v < 8; ++v) {
    u32 lo = base[(size_t)(k0 + (v << 1))     * ld + n];
    u32 hi = base[(size_t)(k0 + (v << 1) + 1) * ld + n];
    r.u[v] = lo | (hi << 16);
  }
  return r.v;
}

// B fragment with per-lane row base (rel-pos gather rows)
__device__ __forceinline__ bf16x16 load_b16_rowptr(const u16* rowbase, int kbase) {
  int lane = threadIdx.x & 31;
  const u16* p = rowbase + kbase + ((lane >> 4) << 4);
  union { bf16x16 v; u32 u[8]; } r;
#pragma unroll
  for (int v = 0; v < 8; ++v) r.u[v] = *(const u32*)(p + (v << 1));
  return r.v;
}

__device__ __forceinline__ f32x8 wmma_bf16(bf16x16 a, bf16x16 b, f32x8 c) {
  return __builtin_amdgcn_wmma_f32_16x16x32_bf16(false, a, false, b, (short)0, c, false, false);
}

// reductions within a 16-lane half of the wave (rows of a C fragment)
__device__ __forceinline__ float hmax16(float x) {
#pragma unroll
  for (int m = 1; m < 16; m <<= 1) x = fmaxf(x, __shfl_xor(x, m, 32));
  return x;
}
__device__ __forceinline__ float hsum16(float x) {
#pragma unroll
  for (int m = 1; m < 16; m <<= 1) x += __shfl_xor(x, m, 32);
  return x;
}

#if ATT_TDM
// Tensor Data Mover: contiguous 1-D tile of n2B bf16 elements, global -> LDS.
// D# per CDNA5 ISA (group0: count/lds/global/type, group1: dims).
__device__ __forceinline__ void tdm_load_1d(void* lds_dst, const void* gsrc, u32 n2B) {
  u32 laddr = (u32)(unsigned long long)lds_dst;           // low 32 bits = LDS byte address
  unsigned long long ga = (unsigned long long)gsrc;
  u32x4 g0;
  g0.x = 1u;                                              // count=1, user descriptor
  g0.y = laddr;                                           // lds_addr (bytes)
  g0.z = (u32)ga;                                         // global_addr[31:0]
  g0.w = (u32)((ga >> 32) & 0x01FFFFFFu) | (2u << 30);    // global_addr[56:32] | type=2
  i32x8 g1;
  g1[0] = (int)(1u << 16);                                // data_size=1 -> 2 bytes/elem
  g1[1] = (int)((n2B & 0xFFFFu) << 16);                   // tensor_dim0[15:0]
  g1[2] = (int)(((n2B >> 16) & 0xFFFFu) | (1u << 16));    // tensor_dim0[31:16], tensor_dim1=1
  g1[3] = (int)((n2B & 0xFFFFu) << 16);                   // tile_dim0 = n2B
  g1[4] = 0;                                              // tile_dim1/2 unused
  g1[5] = (int)n2B;                                       // tensor_dim0_stride
  g1[6] = 0;
  g1[7] = 0;
  i32x4 gz = {0, 0, 0, 0};
#if __clang_major__ >= 23
  i32x8 gz8 = {0, 0, 0, 0, 0, 0, 0, 0};
  __builtin_amdgcn_tensor_load_to_lds(g0, g1, gz, gz, gz8, 0);
#else
  __builtin_amdgcn_tensor_load_to_lds(g0, g1, gz, gz, 0);
#endif
}
#endif

// ---------------- kernel 0: fp32 -> bf16 ----------------

__global__ void cvt_bf16_kernel(const float* __restrict__ in, u16* __restrict__ out, int n) {
  int i = blockIdx.x * 256 + threadIdx.x;
  if (i < n) out[i] = f2bf(in[i]);
}

// ---------------- kernels 1 & 5: GEMM  C = A(bf16) * B(bf16)^T + bias ----------------
// A: MxK row-major bf16; B: NxK row-major bf16; C: MxN fp32 (ldc)
// block = 256 (8 waves); wave tile = 32(M) x 64(N); grid = (M/256, N/64)

__global__ void gemm_bf16_kernel(const u16* __restrict__ A, const u16* __restrict__ B,
                                 const float* __restrict__ bias, float* __restrict__ C,
                                 int K, int ldc) {
  int wave = threadIdx.x >> 5;
  int lane = threadIdx.x & 31;
  int hilf = lane >> 4, nlane = lane & 15;
  int m0 = blockIdx.x * 256 + wave * 32;
  int n0 = blockIdx.y * 64;

  f32x8 z = {0.f,0.f,0.f,0.f,0.f,0.f,0.f,0.f};
  f32x8 acc[2][4];
#pragma unroll
  for (int mh = 0; mh < 2; ++mh)
#pragma unroll
    for (int nb = 0; nb < 4; ++nb) acc[mh][nb] = z;

  for (int k0 = 0; k0 < K; k0 += 32) {
    bf16x16 a0 = load_a16(A, K, m0, k0);
    bf16x16 a1 = load_a16(A, K, m0 + 16, k0);
#pragma unroll
    for (int nb = 0; nb < 4; ++nb) {
      bf16x16 b = load_b16(B, K, n0 + nb * 16, k0);
      acc[0][nb] = wmma_bf16(a0, b, acc[0][nb]);
      acc[1][nb] = wmma_bf16(a1, b, acc[1][nb]);
    }
  }
#pragma unroll
  for (int mh = 0; mh < 2; ++mh) {
#pragma unroll
    for (int nb = 0; nb < 4; ++nb) {
      int n = n0 + nb * 16 + nlane;
      float bv = bias[n];
#pragma unroll
      for (int r = 0; r < 8; ++r) {
        int m = m0 + mh * 16 + r + (hilf << 3);
        C[(size_t)m * ldc + n] = acc[mh][nb][r] + bv;
      }
    }
  }
}

// ---------------- kernel 2: depthwise 3x3 conv + LayerNorm(head_dim) ----------------
// in : qkv_raw (4096 x 2304) fp32, channel n = (t*12+head)*64 + c
// out: P32/P16 [t*12+head][m][c]  (fp32 + bf16)
// grid = (1024, 36), block = (64, 4)

__global__ void dwconv_ln_kernel(const float* __restrict__ qkvraw,
                                 const float* __restrict__ wpq, const float* __restrict__ wpk,
                                 const float* __restrict__ wpv,
                                 const float* __restrict__ gq, const float* __restrict__ bq,
                                 const float* __restrict__ gk, const float* __restrict__ bk,
                                 const float* __restrict__ gv, const float* __restrict__ bv,
                                 float* __restrict__ P32, u16* __restrict__ P16) {
  int th = blockIdx.y;            // t*12 + head
  int t  = th / NH;
  const float* wp = (t == 0) ? wpq : (t == 1) ? wpk : wpv;
  const float* g  = (t == 0) ? gq  : (t == 1) ? gk  : gv;
  const float* b  = (t == 0) ? bq  : (t == 1) ? bk  : bv;

  int c  = threadIdx.x;                       // 0..63
  int ty = threadIdx.y;                       // 0..3
  int m  = blockIdx.x * 4 + ty;               // spatial index
  int h  = m >> 6, w = m & 63;

  float acc = 0.f;
#pragma unroll
  for (int dy = 0; dy < 3; ++dy) {
#pragma unroll
    for (int dx = 0; dx < 3; ++dx) {
      int hh = h + dy - 1, ww = w + dx - 1;
      if (hh >= 0 && hh < 64 && ww >= 0 && ww < 64) {
        acc += qkvraw[(size_t)(hh * 64 + ww) * QKVN + th * HD + c] * wp[(dy * 3 + dx) * HD + c];
      }
    }
  }

  __shared__ float s1[4][64], s2[4][64];
  s1[ty][c] = acc; s2[ty][c] = acc * acc;
  __syncthreads();
#pragma unroll
  for (int s = 32; s > 0; s >>= 1) {
    if (c < s) { s1[ty][c] += s1[ty][c + s]; s2[ty][c] += s2[ty][c + s]; }
    __syncthreads();
  }
  float mean = s1[ty][0] * (1.f / 64.f);
  float var  = s2[ty][0] * (1.f / 64.f) - mean * mean;
  float y = (acc - mean) * rsqrtf(var + 1e-5f) * g[c] + b[c];

  size_t o = ((size_t)th * SEQ + m) * HD + c;
  P32[o] = y;
  P16[o] = f2bf(y);
}

// ---------------- kernel 3: rel-pos bias tensors ----------------
// mode 0: rel_h[head][h][w][kh] = sum_c qhat[head,h,w,c] * rel_pos_h[h-kh+63, c]
// mode 1: rel_w[head][h][w][kw] = sum_c qhat[head,h,w,c] * rel_pos_w[w-kw+63, c]
// grid = (64, 12); block = 128 (4 waves x 16 rows)

__global__ void relpos_kernel(const u16* __restrict__ qhat, const u16* __restrict__ rp,
                              float* __restrict__ out, int mode) {
  int fix  = blockIdx.x;
  int head = blockIdx.y;
  int wave = threadIdx.x >> 5;
  int lane = threadIdx.x & 31;
  int hilf = lane >> 4, nlane = lane & 15;

  const u16* abase; int ald;
  if (mode == 0) { abase = qhat + ((size_t)head * SEQ + fix * 64) * HD; ald = HD; }       // rows = w
  else           { abase = qhat + ((size_t)head * SEQ + fix) * HD;      ald = 64 * HD; }  // rows = h

  f32x8 z = {0.f,0.f,0.f,0.f,0.f,0.f,0.f,0.f};
  f32x8 acc[4];
#pragma unroll
  for (int nb = 0; nb < 4; ++nb) acc[nb] = z;

#pragma unroll
  for (int kb = 0; kb < 2; ++kb) {
    bf16x16 a = load_a16(abase, ald, wave * 16, kb * 32);
#pragma unroll
    for (int nb = 0; nb < 4; ++nb) {
      int n = nb * 16 + nlane;                 // kh / kw
      const u16* rowptr = rp + (size_t)(fix + 63 - n) * HD;
      bf16x16 b = load_b16_rowptr(rowptr, kb * 32);
      acc[nb] = wmma_bf16(a, b, acc[nb]);
    }
  }
#pragma unroll
  for (int nb = 0; nb < 4; ++nb) {
    int n = nb * 16 + nlane;
#pragma unroll
    for (int r = 0; r < 8; ++r) {
      int m = wave * 16 + r + (hilf << 3);
      size_t o = (mode == 0)
        ? (((size_t)(head * 64 + fix) * 64 + m) * 64 + n)     // [head][h=fix][w=m][kh=n]
        : (((size_t)(head * 64 + m) * 64 + fix) * 64 + n);    // [head][h=m][w=fix][kw=n]
      out[o] = acc[nb][r];
    }
  }
}

// ---------------- kernel 4: flash attention + residual ----------------
// grid = (64 query-rows h, 12 heads); block = 128 (4 waves, 16 query rows each)
// K/V tiles (8 KB each) double-buffered in LDS via the Tensor Data Mover.

__global__ void attention_kernel(const u16* __restrict__ P16, const float* __restrict__ P32,
                                 const float* __restrict__ relh, const float* __restrict__ relw,
                                 u16* __restrict__ OB) {
  int h    = blockIdx.x;
  int head = blockIdx.y;
  int tid  = threadIdx.x;
  int wave = tid >> 5;
  int lane = tid & 31;
  int hilf = lane >> 4, nlane = lane & 15;

  const u16* qB = P16 + ((size_t)(        head) * SEQ) * HD;
  const u16* kB = P16 + ((size_t)(NH    + head) * SEQ) * HD;
  const u16* vB = P16 + ((size_t)(2*NH  + head) * SEQ) * HD;

  __shared__ u16 Kt[2][64 * HD];               // double-buffered K tile (bf16)
  __shared__ u16 Vt[2][64 * HD];               // double-buffered V tile (bf16)
  __shared__ u16 Pl[4][16 * 64];               // per-wave P tile (bf16)

  int m0 = h * 64 + wave * 16;                 // global query row base for this wave
  bf16x16 aQ0 = load_a16(qB, HD, m0, 0);
  bf16x16 aQ1 = load_a16(qB, HD, m0, 32);

  f32x8 z = {0.f,0.f,0.f,0.f,0.f,0.f,0.f,0.f};
  f32x8 acc[4];
  float mi[8], li[8];
  size_t rbase[8];
#pragma unroll
  for (int r = 0; r < 8; ++r) {
    mi[r] = -1e30f; li[r] = 0.f;
    int wq = wave * 16 + r + (hilf << 3);      // this row's w coordinate
    rbase[r] = ((size_t)(head * 64 + h) * 64 + wq) * 64;
  }
#pragma unroll
  for (int cb = 0; cb < 4; ++cb) acc[cb] = z;

  // stage tile 0
#if ATT_TDM
  if (wave == 0) {
    tdm_load_1d(&Kt[0][0], kB, 64 * HD);
    tdm_load_1d(&Vt[0][0], vB, 64 * HD);
  }
#else
  {
    const u32* sk = (const u32*)kB; const u32* sv = (const u32*)vB;
    u32* dk = (u32*)&Kt[0][0]; u32* dv = (u32*)&Vt[0][0];
    for (int i = tid; i < (64 * HD) / 2; i += 128) { dk[i] = sk[i]; dv[i] = sv[i]; }
  }
#endif

  for (int j = 0; j < 64; ++j) {               // key tiles: key = j*64 + kw
    int buf = j & 1;
    // stage tile j+1 into the other buffer, then ensure tile j's DMA is done
    if (j < 63) {
#if ATT_TDM
      if (wave == 0) {
        tdm_load_1d(&Kt[buf ^ 1][0], kB + (size_t)(j + 1) * 64 * HD, 64 * HD);
        tdm_load_1d(&Vt[buf ^ 1][0], vB + (size_t)(j + 1) * 64 * HD, 64 * HD);
        __builtin_amdgcn_s_wait_tensorcnt(2);
      }
#else
      const u32* sk = (const u32*)(kB + (size_t)(j + 1) * 64 * HD);
      const u32* sv = (const u32*)(vB + (size_t)(j + 1) * 64 * HD);
      u32* dk = (u32*)&Kt[buf ^ 1][0]; u32* dv = (u32*)&Vt[buf ^ 1][0];
      for (int i = tid; i < (64 * HD) / 2; i += 128) { dk[i] = sk[i]; dv[i] = sv[i]; }
#endif
    } else {
#if ATT_TDM
      if (wave == 0) __builtin_amdgcn_s_wait_tensorcnt(0);
#endif
    }
    __syncthreads();

    const u16* Kbuf = &Kt[buf][0];
    const u16* Vbuf = &Vt[buf][0];

    f32x8 S[4];
#pragma unroll
    for (int nb = 0; nb < 4; ++nb) S[nb] = z;
#pragma unroll
    for (int kb = 0; kb < 2; ++kb) {
      bf16x16 a = (kb == 0) ? aQ0 : aQ1;
#pragma unroll
      for (int nb = 0; nb < 4; ++nb) {
        bf16x16 b = load_b16(Kbuf, HD, nb * 16, kb * 32);
        S[nb] = wmma_bf16(a, b, S[nb]);
      }
    }
    // scale + decomposed rel-pos bias
    float rh[8];
#pragma unroll
    for (int r = 0; r < 8; ++r) rh[r] = relh[rbase[r] + j];
#pragma unroll
    for (int nb = 0; nb < 4; ++nb) {
      int kw = nb * 16 + nlane;
#pragma unroll
      for (int r = 0; r < 8; ++r)
        S[nb][r] = S[nb][r] * 0.125f + rh[r] + relw[rbase[r] + kw];
    }
    // online softmax (per-row, rows live in 16-lane halves)
#pragma unroll
    for (int r = 0; r < 8; ++r) {
      float x = S[0][r];
#pragma unroll
      for (int nb = 1; nb < 4; ++nb) x = fmaxf(x, S[nb][r]);
      x = hmax16(x);
      float mnew  = fmaxf(mi[r], x);
      float alpha = __expf(mi[r] - mnew);
      mi[r] = mnew;
      li[r] *= alpha;
#pragma unroll
      for (int cb = 0; cb < 4; ++cb) acc[cb][r] *= alpha;
    }
#pragma unroll
    for (int nb = 0; nb < 4; ++nb)
#pragma unroll
      for (int r = 0; r < 8; ++r) S[nb][r] = __expf(S[nb][r] - mi[r]);
#pragma unroll
    for (int r = 0; r < 8; ++r) {
      float x = S[0][r] + S[1][r] + S[2][r] + S[3][r];
      li[r] += hsum16(x);
    }
    // relayout P (C-frag) -> LDS -> A-frag
#pragma unroll
    for (int nb = 0; nb < 4; ++nb)
#pragma unroll
      for (int r = 0; r < 8; ++r)
        Pl[wave][(r + (hilf << 3)) * 64 + nb * 16 + nlane] = f2bf(S[nb][r]);
    __syncthreads();
#pragma unroll
    for (int kb = 0; kb < 2; ++kb) {
      bf16x16 ap = load_a16_lds(&Pl[wave][0], kb * 32);
#pragma unroll
      for (int cb = 0; cb < 4; ++cb) {
        bf16x16 bv = load_b16_kmajor(Vbuf, HD, cb * 16, kb * 32);
        acc[cb] = wmma_bf16(ap, bv, acc[cb]);
      }
    }
    __syncthreads();
  }
  // epilogue: normalize + residual (ori_q) -> OB (bf16, row-major 768)
#pragma unroll
  for (int cb = 0; cb < 4; ++cb) {
    int c = cb * 16 + nlane;
#pragma unroll
    for (int r = 0; r < 8; ++r) {
      int mrow = m0 + r + (hilf << 3);
      float val = acc[cb][r] / li[r] + P32[((size_t)head * SEQ + mrow) * HD + c];
      OB[(size_t)mrow * DIM + head * HD + c] = f2bf(val);
    }
  }
}

// ---------------- launch ----------------

extern "C" void kernel_launch(void* const* d_in, const int* in_sizes, int n_in,
                              void* d_out, int out_size, void* d_ws, size_t ws_size,
                              hipStream_t stream) {
  const float* x      = (const float*)d_in[0];
  const float* w_qkv  = (const float*)d_in[1];
  const float* b_qkv  = (const float*)d_in[2];
  const float* wpq    = (const float*)d_in[3];
  const float* wpk    = (const float*)d_in[4];
  const float* wpv    = (const float*)d_in[5];
  const float* gq     = (const float*)d_in[6];
  const float* bq     = (const float*)d_in[7];
  const float* gk     = (const float*)d_in[8];
  const float* bk     = (const float*)d_in[9];
  const float* gv     = (const float*)d_in[10];
  const float* bv     = (const float*)d_in[11];
  const float* rph    = (const float*)d_in[12];
  const float* rpw    = (const float*)d_in[13];
  const float* w_proj = (const float*)d_in[14];
  const float* b_proj = (const float*)d_in[15];

  char* ws = (char*)d_ws;
  size_t off = 0;
  auto alloc = [&](size_t bytes) { size_t r = off; off += (bytes + 255) & ~(size_t)255; return r; };

  u16*   xb      = (u16*)  (ws + alloc((size_t)SEQ * DIM * 2));
  u16*   wqkvb   = (u16*)  (ws + alloc((size_t)QKVN * DIM * 2));
  u16*   wprojb  = (u16*)  (ws + alloc((size_t)DIM * DIM * 2));
  u16*   rphb    = (u16*)  (ws + alloc((size_t)127 * HD * 2));
  u16*   rpwb    = (u16*)  (ws + alloc((size_t)127 * HD * 2));
  float* qkvraw  = (float*)(ws + alloc((size_t)SEQ * QKVN * 4));
  float* P32     = (float*)(ws + alloc((size_t)3 * NH * SEQ * HD * 4));
  u16*   P16     = (u16*)  (ws + alloc((size_t)3 * NH * SEQ * HD * 2));
  float* relH    = (float*)(ws + alloc((size_t)NH * 64 * 64 * 64 * 4));
  float* relW    = (float*)(ws + alloc((size_t)NH * 64 * 64 * 64 * 4));
  u16*   OB      = (u16*)  (ws + alloc((size_t)SEQ * DIM * 2));
  (void)ws_size; (void)n_in; (void)in_sizes; (void)out_size;

  // 0) fp32 -> bf16 conversions
  auto cvt = [&](const float* in, u16* out, int n) {
    cvt_bf16_kernel<<<(n + 255) / 256, 256, 0, stream>>>(in, out, n);
  };
  cvt(x,      xb,     SEQ * DIM);
  cvt(w_qkv,  wqkvb,  QKVN * DIM);
  cvt(w_proj, wprojb, DIM * DIM);
  cvt(rph,    rphb,   127 * HD);
  cvt(rpw,    rpwb,   127 * HD);

  // 1) QKV projection: (4096x768) @ (2304x768)^T + b_qkv
  gemm_bf16_kernel<<<dim3(SEQ / 256, QKVN / 64), 256, 0, stream>>>(
      xb, wqkvb, b_qkv, qkvraw, DIM, QKVN);

  // 2) depthwise 3x3 conv + LayerNorm over head_dim
  dwconv_ln_kernel<<<dim3(SEQ / 4, 3 * NH), dim3(64, 4), 0, stream>>>(
      qkvraw, wpq, wpk, wpv, gq, bq, gk, bk, gv, bv, P32, P16);

  // 3) decomposed rel-pos bias tensors
  relpos_kernel<<<dim3(64, NH), 128, 0, stream>>>(P16, rphb, relH, 0);
  relpos_kernel<<<dim3(64, NH), 128, 0, stream>>>(P16, rpwb, relW, 1);

  // 4) flash attention + residual pooling (TDM-staged K/V)
  attention_kernel<<<dim3(64, NH), 128, 0, stream>>>(P16, P32, relH, relW, OB);

  // 5) output projection: (4096x768) @ (768x768)^T + b_proj -> d_out fp32
  gemm_bf16_kernel<<<dim3(SEQ / 256, DIM / 64), 256, 0, stream>>>(
      OB, wprojb, b_proj, (float*)d_out, DIM, DIM);
}
